// DropCorrelatedAssets_81381040324939
// MI455X (gfx1250) — matrix operand order repaired and checked by reading, not compiled, not run on hardware
//
#include <hip/hip_runtime.h>
#include <hip/hip_bf16.h>

#define BB 64
#define TT 1024
#define NN 512
#define THR 0.15f

typedef __attribute__((ext_vector_type(16))) __bf16 v16bf;
typedef __attribute__((ext_vector_type(8)))  __bf16 v8bf;
typedef __attribute__((ext_vector_type(8)))  float  v8f;

// ---------------------------------------------------------------------------
// Kernel 1: per-column mean + inverse norm in ONE pass over the input.
// diag(cov)[n] = sum_t (x-mu)^2 = sum x^2 - T*mu^2, so no second pass needed.
// Also zeroes the per-row "fired" flags (deterministic re-init every call).
// ---------------------------------------------------------------------------
__global__ void k_stats(const float* __restrict__ in, float* __restrict__ mu,
                        float* __restrict__ rd, int* __restrict__ fired) {
  int idx = blockIdx.x * blockDim.x + threadIdx.x;   // [0, B*N)
  int b = idx >> 9;                                  // N == 512
  int n = idx & (NN - 1);
  const float* p = in + (size_t)b * TT * NN + n;     // consecutive lanes -> consecutive n
  float s = 0.f, s2 = 0.f;
  #pragma unroll 4
  for (int t = 0; t < TT; ++t) {
    float x = p[(size_t)t * NN];
    s += x; s2 += x * x;
  }
  float m   = s * (1.0f / TT);
  float var = s2 - (float)TT * m * m;                // = sum (x-mu)^2
  mu[idx] = m;
  rd[idx] = rsqrtf(var);                             // 1/d
  fired[idx] = 0;
}

// ---------------------------------------------------------------------------
// Kernel 2: center, convert to bf16, transpose [B,T,N] -> [B,N,T] via LDS so
// that GEMM operands are contiguous along K (=T). 32x32 tile per block.
// LDS row stride 34 bf16 (68 B) -> lane bank stride 17, gcd(17,64)=1: no
// bank conflicts on the transposed read.
// ---------------------------------------------------------------------------
__global__ void k_center_t(const float* __restrict__ in,
                           const float* __restrict__ mu,
                           __bf16* __restrict__ xt) {
  __shared__ __bf16 tile[32][34];
  int b  = blockIdx.z;
  int t0 = blockIdx.y * 32;
  int n0 = blockIdx.x * 32;
  int lane = threadIdx.x & 31;
  int grp  = threadIdx.x >> 5;                       // 8 groups of 32
  #pragma unroll
  for (int r = 0; r < 4; ++r) {
    int tl = grp + r * 8, nl = lane;
    float x = in[(size_t)b * TT * NN + (size_t)(t0 + tl) * NN + (n0 + nl)];
    tile[tl][nl] = (__bf16)(x - mu[b * NN + n0 + nl]);
  }
  __syncthreads();
  #pragma unroll
  for (int r = 0; r < 4; ++r) {
    int nl = grp + r * 8, tl = lane;                 // lanes -> consecutive t: 64B bursts
    xt[((size_t)b * NN + (n0 + nl)) * TT + (t0 + tl)] = tile[tl][nl];
  }
}

// ---------------------------------------------------------------------------
// Kernel 3: G = Xc^T * Xc per batch via v_wmma_f32_16x16x32_bf16.
// Each wave computes a 32x32 output tile as a 2x2 grid of 16x16 WMMA tiles:
// 2 A rows + 2 B rows feed 4 WMMAs per K step -> 2 b128 loads per WMMA and
// 4 independent accumulator chains for the scheduler.
// 32x32 super-tiles enumerate the upper triangle: 16*17/2 = 136 per batch;
// sub-diagonal 16-tiles swept in on the diagonal are discarded by the strict
// j>i guard in the epilogue.
//
// Operand packing (ISA 7.12.2, wave32):
//  A (16x32, M=i, K=t): lane L holds M=L&15; VGPR0-3 = 8 consecutive K at
//    offset (L>=16 ? 8 : 0), VGPR4-7 = 8 consecutive K at offset 16+(L>=16?8:0).
//  B (32x16, K=t, N=j): lane L holds N=L&15; same K order as A: 8 K at
//    (L>=16 ? 8 : 0), then 8 K at 16+(L>=16?8:0), 2 per VGPR.
// ---------------------------------------------------------------------------
__global__ __launch_bounds__(256)
void k_corr_gemm(const __bf16* __restrict__ xt, const float* __restrict__ rd,
                 int* __restrict__ fired) {
  const int NSUP = 136;                  // 16*17/2 upper-tri 32x32 super-tiles

  int wave = (blockIdx.x * blockDim.x + threadIdx.x) >> 5;
  int lane = threadIdx.x & 31;
  int b = wave / NSUP;
  int p = wave % NSUP;
  int tii = 0;
  while (p >= 16 - tii) { p -= 16 - tii; ++tii; }    // scalar triangular decode
  int tjj = tii + p;
  int ti0 = 2 * tii;                                 // rows:  ti0, ti0+1 (16-tiles)
  int tj0 = 2 * tjj;                                 // cols:  tj0, tj0+1

  int lo = lane & 15, hi = lane >> 4;
  const __bf16* arow0 = xt + ((size_t)b * NN + ti0 * 16      + lo) * TT;
  const __bf16* arow1 = xt + ((size_t)b * NN + ti0 * 16 + 16 + lo) * TT;
  const __bf16* brow0 = xt + ((size_t)b * NN + tj0 * 16      + lo) * TT;
  const __bf16* brow1 = xt + ((size_t)b * NN + tj0 * 16 + 16 + lo) * TT;

  v8f acc00 = {}, acc01 = {}, acc10 = {}, acc11 = {};
  for (int k0 = 0; k0 < TT; k0 += 32) {
    v8bf a00 = *(const v8bf*)(arow0 + k0 + hi * 8);
    v8bf a01 = *(const v8bf*)(arow0 + k0 + 16 + hi * 8);
    v8bf a10 = *(const v8bf*)(arow1 + k0 + hi * 8);
    v8bf a11 = *(const v8bf*)(arow1 + k0 + 16 + hi * 8);
    v8bf b00 = *(const v8bf*)(brow0 + k0 + hi * 8);
    v8bf b01 = *(const v8bf*)(brow0 + k0 + 16 + hi * 8);
    v8bf b10 = *(const v8bf*)(brow1 + k0 + hi * 8);
    v8bf b11 = *(const v8bf*)(brow1 + k0 + 16 + hi * 8);
    v16bf A0 = __builtin_shufflevector(a00, a01, 0,1,2,3,4,5,6,7,8,9,10,11,12,13,14,15);
    v16bf A1 = __builtin_shufflevector(a10, a11, 0,1,2,3,4,5,6,7,8,9,10,11,12,13,14,15);
    v16bf B0 = __builtin_shufflevector(b00, b01, 0,1,2,3,4,5,6,7,8,9,10,11,12,13,14,15);
    v16bf B1 = __builtin_shufflevector(b10, b11, 0,1,2,3,4,5,6,7,8,9,10,11,12,13,14,15);
    acc00 = __builtin_amdgcn_wmma_f32_16x16x32_bf16(false, A0, false, B0,
                                                    (short)0, acc00, false, false);
    acc01 = __builtin_amdgcn_wmma_f32_16x16x32_bf16(false, A0, false, B1,
                                                    (short)0, acc01, false, false);
    acc10 = __builtin_amdgcn_wmma_f32_16x16x32_bf16(false, A1, false, B0,
                                                    (short)0, acc10, false, false);
    acc11 = __builtin_amdgcn_wmma_f32_16x16x32_bf16(false, A1, false, B1,
                                                    (short)0, acc11, false, false);
  }

  // C/D layout: VGPR k -> M = (hi?8:0)+k, N = lo. rd[i] for k=0..7 are
  // 8 consecutive, 32B-aligned floats -> vector loads.
  v8f rdi0 = *(const v8f*)(rd + b * NN + ti0 * 16      + hi * 8);
  v8f rdi1 = *(const v8f*)(rd + b * NN + ti0 * 16 + 16 + hi * 8);
  int j0 = tj0 * 16 + lo;
  int j1 = j0 + 16;
  float rdj0 = rd[b * NN + j0];
  float rdj1 = rd[b * NN + j1];
  #pragma unroll
  for (int k = 0; k < 8; ++k) {
    int i0 = ti0 * 16 + hi * 8 + k;
    int i1 = i0 + 16;
    bool f0 = ((j0 > i0) && (fabsf(acc00[k] * rdi0[k] * rdj0) > THR)) |
              ((j1 > i0) && (fabsf(acc01[k] * rdi0[k] * rdj1) > THR));
    bool f1 = ((j0 > i1) && (fabsf(acc10[k] * rdi1[k] * rdj0) > THR)) |
              ((j1 > i1) && (fabsf(acc11[k] * rdi1[k] * rdj1) > THR));
    if (f0) atomicOr(&fired[b * NN + i0], 1);        // rarely taken for N(0,1) data
    if (f1) atomicOr(&fired[b * NN + i1], 1);
  }
}

// ---------------------------------------------------------------------------
// Kernel 4: collapse fired flags to the mask. One block per batch (512 thr).
// 0 fired -> ones; 1 fired -> one_hot; >=2 -> zeros.
// ---------------------------------------------------------------------------
__global__ void k_mask(const int* __restrict__ fired, float* __restrict__ out) {
  __shared__ int cnt;
  int b = blockIdx.x, n = threadIdx.x;
  if (n == 0) cnt = 0;
  __syncthreads();
  int f = fired[b * NN + n];
  if (f) atomicAdd(&cnt, 1);
  __syncthreads();
  int c = cnt;
  out[b * NN + n] = (c == 0) ? 1.0f : (c == 1 ? (float)f : 0.0f);
}

// ---------------------------------------------------------------------------
extern "C" void kernel_launch(void* const* d_in, const int* in_sizes, int n_in,
                              void* d_out, int out_size, void* d_ws, size_t ws_size,
                              hipStream_t stream) {
  const float* in = (const float*)d_in[0];
  char* ws = (char*)d_ws;

  __bf16* xt   = (__bf16*)ws;                               // 64 MB: [B,N,T] bf16
  size_t XT_B  = (size_t)BB * NN * TT * sizeof(__bf16);
  float* rd    = (float*)(ws + XT_B);                       // 128 KB
  float* mu    = rd + BB * NN;                              // 128 KB
  int*   fired = (int*)(mu + BB * NN);                      // 128 KB
  float* out   = (float*)d_out;

  k_stats<<<(BB * NN) / 256, 256, 0, stream>>>(in, mu, rd, fired);

  dim3 g2(NN / 32, TT / 32, BB);
  k_center_t<<<g2, 256, 0, stream>>>(in, mu, xt);

  const int NSUP = 136;
  int waves  = BB * NSUP;                                   // 8704 waves
  int blocks = (waves * 32) / 256;                          // 1088 blocks
  k_corr_gemm<<<blocks, 256, 0, stream>>>(xt, rd, fired);

  k_mask<<<BB, NN, 0, stream>>>(fired, out);
}
